// RNN_23733989278364
// MI455X (gfx1250) — matrix operand order repaired and checked by reading, not compiled, not run on hardware
//
#include <hip/hip_runtime.h>
#include <hip/hip_bf16.h>

// ---------------------------------------------------------------------------
// Types for CDNA5 WMMA (wave32): D[16x16,f32] = A[16x32,bf16] x B[32x16,bf16] + C
// ---------------------------------------------------------------------------
typedef __attribute__((ext_vector_type(8)))  __bf16 v8bf;
typedef __attribute__((ext_vector_type(16))) __bf16 v16bf;
typedef __attribute__((ext_vector_type(8)))  float  v8f;

__device__ __forceinline__ v8f wmma_bf16(v16bf a, v16bf b, v8f c) {
  return __builtin_amdgcn_wmma_f32_16x16x32_bf16(false, a, false, b, (short)0, c,
                                                 false, false);
}

__device__ __forceinline__ v8f bcast8(float v) {
  v8f r;
#pragma unroll
  for (int i = 0; i < 8; ++i) r[i] = v;
  return r;
}

__device__ __forceinline__ float sigmoidf_(float x) {
  return 1.0f / (1.0f + __expf(-x));
}

// A-fragment (16x32 bf16, ISA 7.12.2): lane l holds row m=l&15;
// K = (l>>4)*8 + [0..7] then 16 + (l>>4)*8 + [0..7]  -> two contiguous 16B LDS loads.
__device__ __forceinline__ v16bf load_a(const __bf16* src, int stride, int kt, int lane) {
  int m  = lane & 15;
  int hi = lane >> 4;
  const __bf16* p = src + m * stride + kt * 32 + hi * 8;
  v8bf lo = *(const v8bf*)p;
  v8bf hh = *(const v8bf*)(p + 16);
  return __builtin_shufflevector(lo, hh, 0, 1, 2, 3, 4, 5, 6, 7,
                                 8, 9, 10, 11, 12, 13, 14, 15);
}

// B-fragment (32x16 bf16): weight kept in original [N][Kpad] row-major bf16.
// lane l holds column n = n0 + (l&15), K = (l>>4)*8 + [0..7] then 16 + (l>>4)*8 + [0..7]: two 16B loads.
__device__ __forceinline__ v16bf load_b(const __bf16* W, int Kpad, int n0, int kt, int lane) {
  int n = n0 + (lane & 15);
  int k = kt * 32 + (lane >> 4) * 8;
  const __bf16* p = W + (size_t)n * Kpad + k;
  v8bf lo = *(const v8bf*)p;
  v8bf hh = *(const v8bf*)(p + 16);
  return __builtin_shufflevector(lo, hh, 0, 1, 2, 3, 4, 5, 6, 7,
                                 8, 9, 10, 11, 12, 13, 14, 15);
}

// D-fragment store: lane l holds (m, n) = ((l>>4)*8 + r, n0 + (l&15)), r = vgpr 0..7
__device__ __forceinline__ void store_tile(__bf16* dst, int stride, int n0, v8f acc,
                                           bool relu, int lane) {
  int n  = n0 + (lane & 15);
  int mb = (lane >> 4) * 8;
#pragma unroll
  for (int r = 0; r < 8; ++r) {
    float v = acc[r];
    if (relu) v = v > 0.f ? v : 0.f;
    dst[(mb + r) * stride + n] = (__bf16)v;
  }
}

template <int K>
__device__ __forceinline__ v8f fc_tile(const __bf16* A, const __bf16* W,
                                       const float* bias, int n0, int lane) {
  v8f acc = bcast8(bias[n0 + (lane & 15)]);
  for (int kt = 0; kt < K / 32; ++kt) {
    v16bf a = load_a(A, K, kt, lane);
    acc = wmma_bf16(a, load_b(W, K, n0, kt, lane), acc);
  }
  return acc;
}

// ---------------------------------------------------------------------------
// One LSTM layer step for a 16-row batch tile.
//   gates[16][4H] = X[16][KX] @ Wih^T + H[16][H] @ Whh^T + bias
// Wave w owns N-tiles {w+8j}; quadruple q -> i-tile t=w+8q, f/g/o at +H/16 tiles,
// so each wave holds complete (i,f,g,o) columns: nonlinearity stays in registers.
// c kept in registers (D-fragment layout); h written bf16 to ping-pong LDS buffer.
// ---------------------------------------------------------------------------
template <int H, int KX>
__device__ void lstm_layer(const __bf16* xsrc, int xstride,
                           const __bf16* hsrc, __bf16* hdst,
                           const __bf16* Wih, const __bf16* Whh,
                           const float* bias, v8f* cq, int wave, int lane) {
  constexpr int nQ = H / 128;
#pragma unroll
  for (int q = 0; q < nQ; ++q) {
    int tt = wave + 8 * q;
    int nI = tt * 16 + (lane & 15);
    v8f ai = bcast8(bias[nI]);
    v8f af = bcast8(bias[nI + H]);
    v8f ag = bcast8(bias[nI + 2 * H]);
    v8f ao = bcast8(bias[nI + 3 * H]);
    for (int kt = 0; kt < KX / 32; ++kt) {
      v16bf a = load_a(xsrc, xstride, kt, lane);
      ai = wmma_bf16(a, load_b(Wih, KX, tt * 16, kt, lane), ai);
      af = wmma_bf16(a, load_b(Wih, KX, tt * 16 + H, kt, lane), af);
      ag = wmma_bf16(a, load_b(Wih, KX, tt * 16 + 2 * H, kt, lane), ag);
      ao = wmma_bf16(a, load_b(Wih, KX, tt * 16 + 3 * H, kt, lane), ao);
    }
    for (int kt = 0; kt < H / 32; ++kt) {
      v16bf a = load_a(hsrc, H, kt, lane);
      ai = wmma_bf16(a, load_b(Whh, H, tt * 16, kt, lane), ai);
      af = wmma_bf16(a, load_b(Whh, H, tt * 16 + H, kt, lane), af);
      ag = wmma_bf16(a, load_b(Whh, H, tt * 16 + 2 * H, kt, lane), ag);
      ao = wmma_bf16(a, load_b(Whh, H, tt * 16 + 3 * H, kt, lane), ao);
    }
    v8f c = cq[q];
#pragma unroll
    for (int r = 0; r < 8; ++r) {
      float ig = sigmoidf_(ai[r]);
      float fg = sigmoidf_(af[r]);
      float gg = tanhf(ag[r]);
      float og = sigmoidf_(ao[r]);
      float cn = fg * c[r] + ig * gg;
      c[r] = cn;
      float hv = og * tanhf(cn);
      hdst[((lane >> 4) * 8 + r) * H + tt * 16 + (lane & 15)] = (__bf16)hv;
    }
    cq[q] = c;
  }
}

// ---------------------------------------------------------------------------
// Kernel parameters (converted bf16 weights + combined fp32 biases in d_ws)
// ---------------------------------------------------------------------------
struct KParams {
  const float* x;
  float* out;
  const __bf16 *W1i, *W1h, *W2i, *W2h, *W3i, *W3h;
  const __bf16 *W4i, *W4h, *W5i, *W5h, *W6i, *W6h;
  const __bf16 *Wf1, *Wf2, *Wf3, *Wf4, *Wf5, *Wf6;
  const float *b1, *b2, *b3, *b4, *b5, *b6;
  const float *bf1, *bf2, *bf3, *bf4, *bf5, *bf6;
};

// Convert fp32 weight [N][K] -> bf16 [N][Kpad] row-major with zero pad.
__global__ void conv_pad_kernel(const float* __restrict__ src, __bf16* __restrict__ dst,
                                int N, int K, int Kpad, int total) {
  int i = blockIdx.x * 256 + threadIdx.x;
  if (i >= total) return;
  int n = i / Kpad;
  int k = i - n * Kpad;
  float v = (n < N && k < K) ? src[(size_t)n * K + k] : 0.f;
  dst[i] = (__bf16)v;
}

// dst = b1 + (b2 ? b2 : 0), zero-padded to npad.
__global__ void bias_kernel(const float* __restrict__ b1, const float* __restrict__ b2,
                            float* __restrict__ dst, int n, int npad) {
  int i = blockIdx.x * 256 + threadIdx.x;
  if (i >= npad) return;
  float v = 0.f;
  if (i < n) v = b1[i] + (b2 ? b2[i] : 0.f);
  dst[i] = v;
}

// ---------------------------------------------------------------------------
// Persistent per-batch-tile kernel: 256 WGs x 256 threads (8 wave32).
// Each WG owns 16 batch rows for all 80 sequential steps; no global sync.
// ---------------------------------------------------------------------------
__global__ __launch_bounds__(256) void lstm_seq2seq_kernel(KParams P) {
  __shared__ __align__(16) __bf16 xbuf[16 * 32];        // x staged, K padded to 32
  __shared__ __align__(16) __bf16 h1b[2][16 * 128];     // ping-pong hidden states
  __shared__ __align__(16) __bf16 h2b[2][16 * 256];
  __shared__ __align__(16) __bf16 h3b[2][16 * 512];
  __shared__ __align__(16) __bf16 a1[16 * 512];         // fc1 out (relu)
  __shared__ __align__(16) __bf16 a3[16 * 512];         // fc4 out (relu)
  __shared__ __align__(16) __bf16 a2[16 * 256];         // fc2 out (relu)
  __shared__ __align__(16) __bf16 a4[16 * 256];         // fc5 out (relu)

  const int tid  = threadIdx.x;
  const int wave = tid >> 5;
  const int lane = tid & 31;
  const int b0   = blockIdx.x * 16;

  // cell state registers in D-fragment layout
  v8f c1q[1], c2q[2], c3q[4];
  c1q[0] = bcast8(0.f);
#pragma unroll
  for (int i = 0; i < 2; ++i) c2q[i] = bcast8(0.f);
#pragma unroll
  for (int i = 0; i < 4; ++i) c3q[i] = bcast8(0.f);

  for (int i = tid; i < 16 * 32; i += 256) xbuf[i] = (__bf16)0.f;
  for (int i = tid; i < 2 * 16 * 128; i += 256) (&h1b[0][0])[i] = (__bf16)0.f;
  for (int i = tid; i < 2 * 16 * 256; i += 256) (&h2b[0][0])[i] = (__bf16)0.f;
  for (int i = tid; i < 2 * 16 * 512; i += 256) (&h3b[0][0])[i] = (__bf16)0.f;
  __syncthreads();

  int p = 0;

  // ---- encoder: 50 steps ----
  for (int t = 0; t < 50; ++t) {
    {
      int m = tid >> 4, k = tid & 15;  // 256 threads = 16x16 tile
      xbuf[m * 32 + k] = (__bf16)P.x[((size_t)(b0 + m) * 50 + t) * 16 + k];
    }
    __syncthreads();
    lstm_layer<128, 32>(xbuf, 32, h1b[p], h1b[1 - p], P.W1i, P.W1h, P.b1, c1q, wave, lane);
    __syncthreads();
    lstm_layer<256, 128>(h1b[1 - p], 128, h2b[p], h2b[1 - p], P.W2i, P.W2h, P.b2, c2q, wave, lane);
    __syncthreads();
    lstm_layer<512, 256>(h2b[1 - p], 256, h3b[p], h3b[1 - p], P.W3i, P.W3h, P.b3, c3q, wave, lane);
    __syncthreads();
    p ^= 1;
  }

  // ---- decoder: 30 steps (xbuf holds last encoder input; fc3 feeds it back) ----
  for (int s = 0; s < 30; ++s) {
    lstm_layer<128, 32>(xbuf, 32, h1b[p], h1b[1 - p], P.W4i, P.W4h, P.b4, c1q, wave, lane);
    __syncthreads();
    lstm_layer<256, 128>(h1b[1 - p], 128, h2b[p], h2b[1 - p], P.W5i, P.W5h, P.b5, c2q, wave, lane);
    __syncthreads();
    lstm_layer<512, 256>(h2b[1 - p], 256, h3b[p], h3b[1 - p], P.W6i, P.W6h, P.b6, c3q, wave, lane);
    __syncthreads();

    const __bf16* h3n = h3b[1 - p];

    // phase A: fc1 (32 tiles) || fc4 (32 tiles), both read h3n
#pragma unroll
    for (int j = 0; j < 8; ++j) {
      int t = wave + 8 * j;
      if (t < 32) {
        v8f acc = fc_tile<512>(h3n, P.Wf1, P.bf1, t * 16, lane);
        store_tile(a1, 512, t * 16, acc, true, lane);
      } else {
        int u = t - 32;
        v8f acc = fc_tile<512>(h3n, P.Wf4, P.bf4, u * 16, lane);
        store_tile(a3, 512, u * 16, acc, true, lane);
      }
    }
    __syncthreads();

    // phase B: fc2 (16 tiles, reads a1) || fc5 (16 tiles, reads a3)
#pragma unroll
    for (int j = 0; j < 4; ++j) {
      int t = wave + 8 * j;
      if (t < 16) {
        v8f acc = fc_tile<512>(a1, P.Wf2, P.bf2, t * 16, lane);
        store_tile(a2, 256, t * 16, acc, true, lane);
      } else {
        int u = t - 16;
        v8f acc = fc_tile<512>(a3, P.Wf5, P.bf5, u * 16, lane);
        store_tile(a4, 256, u * 16, acc, true, lane);
      }
    }
    __syncthreads();

    // phase C: fc3 (1 tile -> feedback x) on wave 0, fc6 (1 tile -> out) on wave 1
    if (wave == 0) {
      v8f acc = fc_tile<256>(a2, P.Wf3, P.bf3, 0, lane);
      store_tile(xbuf, 32, 0, acc, false, lane);  // cols 16..31 stay zero
    } else if (wave == 1) {
      v8f acc = fc_tile<256>(a4, P.Wf6, P.bf6, 0, lane);
      int n = lane & 15, mb = (lane >> 4) * 8;
      if (n < 2) {
#pragma unroll
        for (int r = 0; r < 8; ++r)
          P.out[(size_t)(b0 + mb + r) * 60 + s * 2 + n] = acc[r];
      }
    }
    __syncthreads();
    p ^= 1;
  }
}

// ---------------------------------------------------------------------------
// Host launch: convert weights into d_ws, then run the persistent kernel.
// ---------------------------------------------------------------------------
extern "C" void kernel_launch(void* const* d_in, const int* in_sizes, int n_in,
                              void* d_out, int out_size, void* d_ws, size_t ws_size,
                              hipStream_t stream) {
  (void)in_sizes; (void)n_in; (void)out_size; (void)ws_size;

  char* ws = (char*)d_ws;
  size_t off = 0;
  auto wsalloc = [&](size_t bytes) -> void* {
    off = (off + 255) & ~(size_t)255;
    void* ptr = ws + off;
    off += bytes;
    return ptr;
  };
  auto conv = [&](const void* src, int N, int K, int Npad, int Kpad) -> const __bf16* {
    __bf16* dst = (__bf16*)wsalloc((size_t)Npad * Kpad * sizeof(__bf16));
    int total = Npad * Kpad;
    conv_pad_kernel<<<(total + 255) / 256, 256, 0, stream>>>(
        (const float*)src, dst, N, K, Kpad, total);
    return dst;
  };
  auto bsum = [&](const void* s1, const void* s2, int n, int npad) -> const float* {
    float* dst = (float*)wsalloc((size_t)npad * sizeof(float));
    bias_kernel<<<(npad + 255) / 256, 256, 0, stream>>>(
        (const float*)s1, (const float*)s2, dst, n, npad);
    return dst;
  };

  KParams P;
  P.x   = (const float*)d_in[0];
  P.out = (float*)d_out;

  // lstm1..lstm3 (encoder), lstm4..lstm6 (decoder): Wih, Whh, bih, bhh
  P.W1i = conv(d_in[1], 512, 16, 512, 32);       // K padded 16->32
  P.W1h = conv(d_in[2], 512, 128, 512, 128);
  P.b1  = bsum(d_in[3], d_in[4], 512, 512);
  P.W2i = conv(d_in[5], 1024, 128, 1024, 128);
  P.W2h = conv(d_in[6], 1024, 256, 1024, 256);
  P.b2  = bsum(d_in[7], d_in[8], 1024, 1024);
  P.W3i = conv(d_in[9], 2048, 256, 2048, 256);
  P.W3h = conv(d_in[10], 2048, 512, 2048, 512);
  P.b3  = bsum(d_in[11], d_in[12], 2048, 2048);
  P.W4i = conv(d_in[13], 512, 16, 512, 32);
  P.W4h = conv(d_in[14], 512, 128, 512, 128);
  P.b4  = bsum(d_in[15], d_in[16], 512, 512);
  P.W5i = conv(d_in[17], 1024, 128, 1024, 128);
  P.W5h = conv(d_in[18], 1024, 256, 1024, 256);
  P.b5  = bsum(d_in[19], d_in[20], 1024, 1024);
  P.W6i = conv(d_in[21], 2048, 256, 2048, 256);
  P.W6h = conv(d_in[22], 2048, 512, 2048, 512);
  P.b6  = bsum(d_in[23], d_in[24], 2048, 2048);
  // fc1..fc6
  P.Wf1 = conv(d_in[25], 512, 512, 512, 512);
  P.bf1 = bsum(d_in[26], nullptr, 512, 512);
  P.Wf2 = conv(d_in[27], 256, 512, 256, 512);
  P.bf2 = bsum(d_in[28], nullptr, 256, 256);
  P.Wf3 = conv(d_in[29], 16, 256, 16, 256);
  P.bf3 = bsum(d_in[30], nullptr, 16, 16);
  P.Wf4 = conv(d_in[31], 512, 512, 512, 512);
  P.bf4 = bsum(d_in[32], nullptr, 512, 512);
  P.Wf5 = conv(d_in[33], 256, 512, 256, 512);
  P.bf5 = bsum(d_in[34], nullptr, 256, 256);
  P.Wf6 = conv(d_in[35], 2, 256, 16, 256);       // N padded 2->16
  P.bf6 = bsum(d_in[36], nullptr, 2, 16);

  lstm_seq2seq_kernel<<<4096 / 16, 256, 0, stream>>>(P);
}
